// Encoder_47244640256347
// MI455X (gfx1250) — compile-verified, matrix operand order checked by reading, not run on hardware
//
#include <hip/hip_runtime.h>
#include <hip/hip_bf16.h>
#include <math.h>

// ---------------------------------------------------------------------------
// Types for CDNA5 WMMA (wave32)
// ---------------------------------------------------------------------------
typedef __bf16 bf16_t;
typedef __attribute__((ext_vector_type(16))) __bf16 v16bf;
typedef __attribute__((ext_vector_type(8)))  __bf16 bf16x8;
typedef __attribute__((ext_vector_type(8)))  float  v8f;

#define TILE_M 256
#define TILE_N 128
#define TILE_K 32
#define LDS_STRIDE (TILE_K + 8)   // 40 bf16 = 80B row stride (16B aligned)

#define GEMM_MODE_BF16       0    // C -> bf16, no bias
#define GEMM_MODE_RELU_BF16  1    // C -> relu(C + bias) bf16
#define GEMM_MODE_BIAS_F32   2    // C -> (C + bias) f32

// ---------------------------------------------------------------------------
// CDNA5 async global->LDS path (ASYNCcnt), with sync fallback.
// Builtin signature (from hipcc diagnostic):
//   void __builtin_amdgcn_global_load_async_to_lds_b128(
//       v4i __device__(AS1)* src, v4i AS3* dst, imm int offset, imm int cpol)
// ---------------------------------------------------------------------------
#if defined(__has_builtin)
#if __has_builtin(__builtin_amdgcn_global_load_async_to_lds_b128)
#define USE_ASYNC_LDS 1
#endif
#endif
#ifndef USE_ASYNC_LDS
#define USE_ASYNC_LDS 0
#endif

typedef int v4i_t __attribute__((vector_size(4 * sizeof(int))));
typedef __attribute__((address_space(1))) v4i_t gbl_v4i;
typedef __attribute__((address_space(3))) v4i_t lds_v4i;

__device__ __forceinline__ gbl_v4i* to_gbl(const void* p) {
  return (gbl_v4i*)(unsigned long long)p;
}
__device__ __forceinline__ lds_v4i* to_lds(void* p) {
  // Generic LDS pointers carry the LDS offset in the low 32 bits.
  return (lds_v4i*)(unsigned int)(unsigned long long)p;
}

__device__ __forceinline__ void async_wait0() {
#if USE_ASYNC_LDS
#if __has_builtin(__builtin_amdgcn_s_wait_asynccnt)
  __builtin_amdgcn_s_wait_asynccnt(0);
#else
  asm volatile("s_wait_asynccnt 0x0" ::: "memory");
#endif
#endif
}

// ---------------------------------------------------------------------------
// Band-scale vector: scale[f] for f in [0, N)
// ---------------------------------------------------------------------------
__global__ void gen_scale_kernel(const float* __restrict__ fw,
                                 const float* __restrict__ kptr,
                                 float* __restrict__ scale, int Nn) {
  int f = blockIdx.x * blockDim.x + threadIdx.x;
  if (f >= Nn) return;
  float kv = kptr[0];
  float sc = 1.0f;
  for (int i = 0; i < 30; ++i) {
    int s = (int)((double)(Nn - 1) * (double)i       / 30.0);
    int e = (int)((double)(Nn - 1) * (double)(i + 1) / 30.0);
    if (f >= s && f < e) {
      if (e <= 30) sc = 1.0f + fw[i] * kv * (1.0f - (float)i / 30.0f);
      else         sc = 1.0f - fw[i] * kv * (1.0f - (float)(i - 30) / 30.0f);
    }
  }
  scale[f] = sc;
}

// ---------------------------------------------------------------------------
// DCT matrices, pre-transposed so every GEMM is C = A * B^T:
//   DT[n][k]   = D[k][n]            (orthonormal DCT-II)
//   DscT[n][k] = D[k][n] * scale[k]
// Then M = DscT * DT^T = D^T diag(scale) D   (symmetric).
// ---------------------------------------------------------------------------
__global__ void gen_dct_kernel(bf16_t* __restrict__ DT,
                               bf16_t* __restrict__ DscT,
                               const float* __restrict__ scale, int Nn) {
  long idx = (long)blockIdx.x * blockDim.x + threadIdx.x;
  long total = (long)Nn * Nn;
  if (idx >= total) return;
  int r  = (int)(idx / Nn);   // spatial index n
  int kk = (int)(idx % Nn);   // frequency index k
  double s = (kk == 0) ? sqrt(1.0 / (double)Nn) : sqrt(2.0 / (double)Nn);
  double v = s * cos(M_PI * ((double)r + 0.5) * (double)kk / (double)Nn);
  float vf = (float)v;
  DT[idx]   = (bf16_t)vf;
  DscT[idx] = (bf16_t)(vf * scale[kk]);
}

// ---------------------------------------------------------------------------
// f32 -> bf16 convert
// ---------------------------------------------------------------------------
__global__ void cvt_f32_bf16_kernel(const float* __restrict__ in,
                                    bf16_t* __restrict__ out, long n) {
  long i = ((long)blockIdx.x * blockDim.x + threadIdx.x) * 4;
  if (i + 3 < n) {
    float4 v = *(const float4*)(in + i);
    out[i + 0] = (bf16_t)v.x;
    out[i + 1] = (bf16_t)v.y;
    out[i + 2] = (bf16_t)v.z;
    out[i + 3] = (bf16_t)v.w;
  } else {
    for (long j = i; j < n; ++j) out[j] = (bf16_t)in[j];
  }
}

// ---------------------------------------------------------------------------
// WMMA GEMM:  C[M,N] = A[M,K] * B[N,K]^T   (both row-major, K-contiguous)
// 256x128 block tile, 8 wave32s in a 4(M) x 2(N) grid; each wave owns a
// 64x64 sub-tile = 4x4 WMMA 16x16x32 bf16 tiles (128 f32 acc VGPRs).
// Global->LDS staging uses GLOBAL_LOAD_ASYNC_TO_LDS_B128 when available.
// ---------------------------------------------------------------------------
__global__ __launch_bounds__(256)
void gemm_nt_bf16_kernel(const bf16_t* __restrict__ A,
                         const bf16_t* __restrict__ B,
                         const float* __restrict__ bias,
                         void* __restrict__ Cout,
                         int M, int N, int K, int mode) {
  __shared__ bf16_t As[2][TILE_M][LDS_STRIDE];
  __shared__ bf16_t Bs[2][TILE_N][LDS_STRIDE];

  const int t    = threadIdx.x;
  const int lane = t & 31;
  const int wave = t >> 5;
  const int bm0  = blockIdx.y * TILE_M;
  const int bn0  = blockIdx.x * TILE_N;

  const int wm0 = (wave & 3) * 64;   // wave M origin within block tile
  const int wn0 = (wave >> 2) * 64;  // wave N origin within block tile

  // CDNA5 16-bit A/B operand layout: lanes 0-15 hold row (lane&15) with
  // K = {0..7, 16..23}; lanes 16-31 hold K = {8..15, 24..31}.
  const int lrow = lane & 15;
  const int kh   = (lane >> 4) * 8;

  v8f acc[4][4];
  const v8f vzero = {0.f, 0.f, 0.f, 0.f, 0.f, 0.f, 0.f, 0.f};
#pragma unroll
  for (int i = 0; i < 4; ++i)
#pragma unroll
    for (int j = 0; j < 4; ++j) acc[i][j] = vzero;

  // Staging: K-slab row = 32 bf16 = 64B = 4 x b128 chunks.
  // A: 256 rows * 4 chunks = 1024 chunks; B: 128 * 4 = 512 chunks.
  // Thread t covers A chunks {t, t+256, t+512, t+768}, B chunks {t, t+256}.
  auto stage = [&](int buf, int kk) {
#if USE_ASYNC_LDS
#pragma unroll
    for (int i = 0; i < 4; ++i) {
      const int c = t + i * 256;
      const int row = c >> 2, col = (c & 3) * 8;
      __builtin_amdgcn_global_load_async_to_lds_b128(
          to_gbl(A + (size_t)(bm0 + row) * K + kk + col),
          to_lds(&As[buf][row][col]), 0, 0);
    }
#pragma unroll
    for (int i = 0; i < 2; ++i) {
      const int c = t + i * 256;
      const int row = c >> 2, col = (c & 3) * 8;
      __builtin_amdgcn_global_load_async_to_lds_b128(
          to_gbl(B + (size_t)(bn0 + row) * K + kk + col),
          to_lds(&Bs[buf][row][col]), 0, 0);
    }
#else
#pragma unroll
    for (int i = 0; i < 4; ++i) {
      const int c = t + i * 256;
      const int row = c >> 2, col = (c & 3) * 8;
      bf16x8 v = *(const bf16x8*)(A + (size_t)(bm0 + row) * K + kk + col);
      *(bf16x8*)&As[buf][row][col] = v;
    }
#pragma unroll
    for (int i = 0; i < 2; ++i) {
      const int c = t + i * 256;
      const int row = c >> 2, col = (c & 3) * 8;
      bf16x8 v = *(const bf16x8*)(B + (size_t)(bn0 + row) * K + kk + col);
      *(bf16x8*)&Bs[buf][row][col] = v;
    }
#endif
  };

  stage(0, 0);
  async_wait0();
  __syncthreads();

  int buf = 0;
  for (int kk = 0; kk < K; kk += TILE_K) {
    if (kk + TILE_K < K) stage(buf ^ 1, kk + TILE_K);

    v16bf afrag[4], bfrag[4];
#pragma unroll
    for (int i = 0; i < 4; ++i) {
      const bf16_t* p = &As[buf][wm0 + i * 16 + lrow][kh];
      bf16x8 lo = *(const bf16x8*)p;
      bf16x8 hi = *(const bf16x8*)(p + 16);
      afrag[i] = __builtin_shufflevector(lo, hi, 0, 1, 2, 3, 4, 5, 6, 7,
                                         8, 9, 10, 11, 12, 13, 14, 15);
    }
#pragma unroll
    for (int j = 0; j < 4; ++j) {
      const bf16_t* p = &Bs[buf][wn0 + j * 16 + lrow][kh];
      bf16x8 lo = *(const bf16x8*)p;
      bf16x8 hi = *(const bf16x8*)(p + 16);
      bfrag[j] = __builtin_shufflevector(lo, hi, 0, 1, 2, 3, 4, 5, 6, 7,
                                         8, 9, 10, 11, 12, 13, 14, 15);
    }

#pragma unroll
    for (int i = 0; i < 4; ++i)
#pragma unroll
      for (int j = 0; j < 4; ++j)
        acc[i][j] = __builtin_amdgcn_wmma_f32_16x16x32_bf16(
            false, afrag[i], false, bfrag[j], (short)0, acc[i][j],
            false, false);

    async_wait0();
    __syncthreads();
    buf ^= 1;
  }

  // Epilogue. C/D layout: VGPR v, lanes 0-15 -> (M=v, N=lane);
  // lanes 16-31 -> (M=v+8, N=lane-16).
  const int cn = lane & 15;
  const int cm = (lane >> 4) * 8;
#pragma unroll
  for (int i = 0; i < 4; ++i) {
#pragma unroll
    for (int j = 0; j < 4; ++j) {
      const int col = bn0 + wn0 + j * 16 + cn;
      const float bval = (mode != GEMM_MODE_BF16) ? bias[col] : 0.f;
#pragma unroll
      for (int v = 0; v < 8; ++v) {
        const int row = bm0 + wm0 + i * 16 + cm + v;
        float val = acc[i][j][v] + bval;
        if (mode == GEMM_MODE_RELU_BF16) val = fmaxf(val, 0.f);
        if (mode == GEMM_MODE_BIAS_F32)
          ((float*)Cout)[(size_t)row * N + col] = val;
        else
          ((bf16_t*)Cout)[(size_t)row * N + col] = (bf16_t)val;
      }
    }
  }
}

// ---------------------------------------------------------------------------
// Host side
// ---------------------------------------------------------------------------
static void launch_gemm(const bf16_t* A, const bf16_t* B, const float* bias,
                        void* C, int M, int N, int K, int mode,
                        hipStream_t s) {
  dim3 grid(N / TILE_N, M / TILE_M);
  gemm_nt_bf16_kernel<<<grid, dim3(256), 0, s>>>(A, B, bias, C, M, N, K, mode);
}

static void launch_cvt(const float* in, bf16_t* out, long n, hipStream_t s) {
  long quads = (n + 3) / 4;
  int blocks = (int)((quads + 255) / 256);
  cvt_f32_bf16_kernel<<<blocks, 256, 0, s>>>(in, out, n);
}

extern "C" void kernel_launch(void* const* d_in, const int* in_sizes, int n_in,
                              void* d_out, int out_size, void* d_ws,
                              size_t ws_size, hipStream_t stream) {
  const int DIM = 2048, BATCH = 16384;
  const int H1 = 1024, H2 = 512, HO = 256;

  const float* x   = (const float*)d_in[0];
  const float* fw  = (const float*)d_in[1];
  const float* kpt = (const float*)d_in[2];
  const float* W1  = (const float*)d_in[3];
  const float* b1  = (const float*)d_in[4];
  const float* W2  = (const float*)d_in[5];
  const float* b2  = (const float*)d_in[6];
  const float* Wmu = (const float*)d_in[7];
  const float* bmu = (const float*)d_in[8];
  const float* Wlv = (const float*)d_in[9];
  const float* blv = (const float*)d_in[10];
  float* out = (float*)d_out;

  char* ws = (char*)d_ws;
  // Workspace layout (bytes); h1 reuses the x_bf16 region, h2 reuses DT+DscT.
  const size_t OFF_XB    = 0;                                        // 64 MB
  const size_t OFF_XE    = OFF_XB    + (size_t)BATCH * DIM * 2;      // 64 MB
  const size_t OFF_DT    = OFF_XE    + (size_t)BATCH * DIM * 2;      // 8 MB
  const size_t OFF_DSCT  = OFF_DT    + (size_t)DIM * DIM * 2;        // 8 MB
  const size_t OFF_MM    = OFF_DSCT  + (size_t)DIM * DIM * 2;        // 8 MB
  const size_t OFF_W1B   = OFF_MM    + (size_t)DIM * DIM * 2;        // 4 MB
  const size_t OFF_W2B   = OFF_W1B   + (size_t)H1 * DIM * 2;         // 1 MB
  const size_t OFF_WMUB  = OFF_W2B   + (size_t)H2 * H1 * 2;          // 256 KB
  const size_t OFF_WLVB  = OFF_WMUB  + (size_t)HO * H2 * 2;          // 256 KB
  const size_t OFF_SCALE = OFF_WLVB  + (size_t)HO * H2 * 2;          // 8 KB

  bf16_t* xB    = (bf16_t*)(ws + OFF_XB);
  bf16_t* xeB   = (bf16_t*)(ws + OFF_XE);
  bf16_t* DT    = (bf16_t*)(ws + OFF_DT);
  bf16_t* DscT  = (bf16_t*)(ws + OFF_DSCT);
  bf16_t* Mm    = (bf16_t*)(ws + OFF_MM);
  bf16_t* W1B   = (bf16_t*)(ws + OFF_W1B);
  bf16_t* W2B   = (bf16_t*)(ws + OFF_W2B);
  bf16_t* WmuB  = (bf16_t*)(ws + OFF_WMUB);
  bf16_t* WlvB  = (bf16_t*)(ws + OFF_WLVB);
  float*  scale = (float*)(ws + OFF_SCALE);
  bf16_t* h1B   = (bf16_t*)(ws + OFF_XB);   // reuse after xe is built
  bf16_t* h2B   = (bf16_t*)(ws + OFF_DT);   // reuse after Mm is built

  // 1) band scale + DCT matrices
  gen_scale_kernel<<<(DIM + 255) / 256, 256, 0, stream>>>(fw, kpt, scale, DIM);
  {
    long total = (long)DIM * DIM;
    gen_dct_kernel<<<(int)((total + 255) / 256), 256, 0, stream>>>(DT, DscT,
                                                                   scale, DIM);
  }

  // 2) converts
  launch_cvt(x,   xB,   (long)BATCH * DIM, stream);
  launch_cvt(W1,  W1B,  (long)H1 * DIM,    stream);
  launch_cvt(W2,  W2B,  (long)H2 * H1,     stream);
  launch_cvt(Wmu, WmuB, (long)HO * H2,     stream);
  launch_cvt(Wlv, WlvB, (long)HO * H2,     stream);

  // 3) Mm = DscT * DT^T = D^T diag(scale) D  (symmetric)
  launch_gemm(DscT, DT, nullptr, Mm, DIM, DIM, DIM, GEMM_MODE_BF16, stream);

  // 4) xe = x * Mm  (Mm symmetric => NT form with Mm as [N,K])
  launch_gemm(xB, Mm, nullptr, xeB, BATCH, DIM, DIM, GEMM_MODE_BF16, stream);

  // 5) h1 = relu(xe * W1^T + b1)
  launch_gemm(xeB, W1B, b1, h1B, BATCH, H1, DIM, GEMM_MODE_RELU_BF16, stream);

  // 6) h2 = relu(h1 * W2^T + b2)
  launch_gemm(h1B, W2B, b2, h2B, BATCH, H2, H1, GEMM_MODE_RELU_BF16, stream);

  // 7) mu, logvar -> d_out (mu first, then logvar, concatenated flat)
  launch_gemm(h2B, WmuB, bmu, out, BATCH, HO, H2, GEMM_MODE_BIAS_F32, stream);
  launch_gemm(h2B, WlvB, blv, out + (size_t)BATCH * HO, BATCH, HO, H2,
              GEMM_MODE_BIAS_F32, stream);

  (void)in_sizes; (void)n_in; (void)out_size; (void)ws_size;
}
